// Qwen3CandidateSelectionAttention_24240795419151
// MI455X (gfx1250) — matrix operand order repaired and checked by reading, not compiled, not run on hardware
//
#include <hip/hip_runtime.h>
#include <hip/hip_bf16.h>

// ---------- problem constants ----------
#define BB 2
#define SS 2048
#define HIDD 2048
#define NHH 16
#define NKVV 8
#define HDD 128
#define SCALING_ 0.08838834764831845f   // 128^-0.5
#define NEG_BIG -1000000000.0f

typedef __attribute__((ext_vector_type(16))) __bf16 v16bf;
typedef __attribute__((ext_vector_type(8)))  float  v8f;
typedef __attribute__((ext_vector_type(4)))  float  f32x4;
typedef __attribute__((ext_vector_type(4)))  unsigned int u32x4;
typedef __attribute__((ext_vector_type(8)))  unsigned int u32x8;

union FragAB { v16bf v; uint4 q[2]; };
union Pack8  { uint4 u; __bf16 h[8]; };

static __device__ inline v8f vzero8() {
  v8f v;
#pragma unroll
  for (int i = 0; i < 8; ++i) v[i] = 0.0f;
  return v;
}

static __device__ inline v8f wmma_bf16(const FragAB& a, const FragAB& b, v8f c) {
  return __builtin_amdgcn_wmma_f32_16x16x32_bf16(false, a.v, false, b.v,
                                                 (short)0, c, false, false);
}

static __device__ inline void wait_tensorcnt0() {
#if defined(__has_builtin)
#if __has_builtin(__builtin_amdgcn_s_wait_tensorcnt)
  __builtin_amdgcn_s_wait_tensorcnt((short)0);
#else
  asm volatile("s_wait_tensorcnt 0x0" ::: "memory");
#endif
#else
  asm volatile("s_wait_tensorcnt 0x0" ::: "memory");
#endif
}

// ---------------- f32 -> bf16 convert (vectorized) ----------------
__global__ __launch_bounds__(256) void cvt_bf16(const float* __restrict__ in,
                                                __bf16* __restrict__ out, int n4) {
  int i = blockIdx.x * 256 + threadIdx.x;
  if (i >= n4) return;
  float4 f = ((const float4*)in)[i];
  union { uint2 u; __bf16 h[4]; } p;
  p.h[0] = (__bf16)f.x; p.h[1] = (__bf16)f.y;
  p.h[2] = (__bf16)f.z; p.h[3] = (__bf16)f.w;
  ((uint2*)out)[i] = p.u;
}

// ---------------- bf16 WMMA GEMM: C[MxN] = A[MxK] @ B[KxN] ----------------
// Block: 256 thr = 8 waves as 2(M) x 4(N); wave tile 32x64; block tile 64x256.
// A tile (64x32, row-major, 80B padded rows) is staged into LDS by the
// Tensor Data Mover (pad_interval = 16 DWORDs, pad_amount = 4 DWORDs).
template <bool OUTF32>
__global__ __launch_bounds__(256) void gemm_bf16(const __bf16* __restrict__ A,
                                                 const __bf16* __restrict__ Bm,
                                                 void* __restrict__ Cout,
                                                 int M, int N, int K) {
  __shared__ alignas(16) __bf16 sA[64][40];    // row stride 80B (16B-multiple)
  __shared__ alignas(16) __bf16 sB[256][40];   // stored transposed: [n][k]
  const int tid = threadIdx.x;
  const int m0 = blockIdx.y * 64;
  const int n0 = blockIdx.x * 256;
  const int w = tid >> 5, lane = tid & 31, lm = lane & 15;
  const bool hi = lane >= 16;
  const int mw = (w >> 2) * 32, nw = (w & 3) * 64;
  const int kbA = hi ? 16 : 0, kbB = hi ? 16 : 0;

  // TDM descriptor group 1 (tile shape / strides / LDS padding), K-invariant.
  u32x8 g1;
  g1[0] = 0x06D10000u;              // data_size=2B, pad_en, intv=16DW, amt=4DW
  g1[1] = (unsigned)(K & 0xFFFF) << 16;                       // tensor_dim0 lo
  g1[2] = (unsigned)((K >> 16) & 0xFFFF) | ((unsigned)(M & 0xFFFF) << 16);
  g1[3] = (unsigned)((M >> 16) & 0xFFFF) | (32u << 16);       // tile_dim0 = 32
  g1[4] = 64u;                      // tile_dim1 = 64 rows
  g1[5] = (unsigned)K;              // tensor_dim0_stride
  g1[6] = 0u;
  g1[7] = 0u;
  const unsigned ldsA = (unsigned)(unsigned long long)(void*)&sA[0][0];

  v8f acc[2][4];
#pragma unroll
  for (int mt = 0; mt < 2; ++mt)
#pragma unroll
    for (int nt = 0; nt < 4; ++nt) acc[mt][nt] = vzero8();

  for (int k0 = 0; k0 < K; k0 += 32) {
    __syncthreads();
    if (w == 0) { // wave 0 drives the TDM for the A tile
      unsigned long long ga = (unsigned long long)(A + (size_t)m0 * K + k0);
      u32x4 g0;
      g0[0] = 1u;                                        // count=1
      g0[1] = ldsA;                                      // lds_addr
      g0[2] = (unsigned)ga;                              // global_addr[31:0]
      g0[3] = (unsigned)((ga >> 32) & 0x1FFFFFFu) | 0x80000000u; // addr[56:32]|type=2
      asm volatile("tensor_load_to_lds %0, %1" :: "s"(g0), "s"(g1) : "memory");
    }
#pragma unroll
    for (int c = 0; c < 4; ++c) { // B tile 32x256, store transposed
      int cid = tid + 256 * c;
      int kk = cid >> 5, n8 = (cid & 31) * 8;
      Pack8 t; t.u = *(const uint4*)(Bm + (size_t)(k0 + kk) * N + n0 + n8);
#pragma unroll
      for (int j = 0; j < 8; ++j) sB[n8 + j][kk] = t.h[j];
    }
    if (w == 0) wait_tensorcnt0();
    __syncthreads();

    FragAB a[2], b[4];
#pragma unroll
    for (int mt = 0; mt < 2; ++mt) {
      a[mt].q[0] = *(const uint4*)&sA[mw + mt * 16 + lm][kbA];
      a[mt].q[1] = *(const uint4*)&sA[mw + mt * 16 + lm][kbA + 8];
    }
#pragma unroll
    for (int nt = 0; nt < 4; ++nt) {
      b[nt].q[0] = *(const uint4*)&sB[nw + nt * 16 + lm][kbB];
      b[nt].q[1] = *(const uint4*)&sB[nw + nt * 16 + lm][kbB + 8];
    }
#pragma unroll
    for (int mt = 0; mt < 2; ++mt)
#pragma unroll
      for (int nt = 0; nt < 4; ++nt)
        acc[mt][nt] = wmma_bf16(a[mt], b[nt], acc[mt][nt]);
  }

#pragma unroll
  for (int mt = 0; mt < 2; ++mt)
#pragma unroll
    for (int nt = 0; nt < 4; ++nt)
#pragma unroll
      for (int r = 0; r < 8; ++r) {
        int row = m0 + mw + mt * 16 + r + (hi ? 8 : 0);
        int col = n0 + nw + nt * 16 + lm;
        if (OUTF32)
          ((float*)Cout)[(size_t)row * N + col] = acc[mt][nt][r];
        else
          ((__bf16*)Cout)[(size_t)row * N + col] = (__bf16)acc[mt][nt][r];
      }
}

// ---------------- RMSNorm + RoPE (in place on bf16 [b,s,nh,HD]) -------------
__global__ __launch_bounds__(256) void norm_rope(__bf16* __restrict__ x,
                                                 const float* __restrict__ cs,
                                                 const float* __restrict__ sn,
                                                 const float* __restrict__ wgt,
                                                 int NHt, float scale) {
  int g = blockIdx.x * 8 + (threadIdx.x >> 5);
  int lane = threadIdx.x & 31;
  int h = g % NHt;
  int s = (g / NHt) % SS;
  int b = g / (NHt * SS);
  __bf16* p = x + ((size_t)(b * SS + s) * NHt + h) * HDD;
  int d0 = lane * 2;
  float a0 = (float)p[d0],      a1 = (float)p[d0 + 1];
  float b0 = (float)p[d0 + 64], b1 = (float)p[d0 + 65];
  float ssq = a0 * a0 + a1 * a1 + b0 * b0 + b1 * b1;
#pragma unroll
  for (int off = 16; off >= 1; off >>= 1) ssq += __shfl_xor(ssq, off, 32);
  float r = rsqrtf(ssq * (1.0f / 128.0f) + 1e-6f);
  a0 *= r * wgt[d0];      a1 *= r * wgt[d0 + 1];
  b0 *= r * wgt[d0 + 64]; b1 *= r * wgt[d0 + 65];
  const float* cp = cs + ((size_t)(b * SS) + s) * HDD;
  const float* sp = sn + ((size_t)(b * SS) + s) * HDD;
  float c0 = cp[d0], c1 = cp[d0 + 1], c2 = cp[d0 + 64], c3 = cp[d0 + 65];
  float s0 = sp[d0], s1 = sp[d0 + 1], s2 = sp[d0 + 64], s3 = sp[d0 + 65];
  p[d0]      = (__bf16)((a0 * c0 - b0 * s0) * scale);
  p[d0 + 1]  = (__bf16)((a1 * c1 - b1 * s1) * scale);
  p[d0 + 64] = (__bf16)((b0 * c2 + a0 * s2) * scale);
  p[d0 + 65] = (__bf16)((b1 * c3 + a1 * s3) * scale);
}

// ---------------- V transpose: [b,s,kv,d] -> [b,kv,d,s] ----------------
__global__ __launch_bounds__(256) void transpose_v(const __bf16* __restrict__ in,
                                                   __bf16* __restrict__ out) {
  int idx = blockIdx.x * 256 + threadIdx.x;           // out-linear index
  int s = idx & (SS - 1);
  int rest = idx >> 11;
  int d = rest & (HDD - 1); rest >>= 7;
  int kv = rest & (NKVV - 1);
  int b = rest >> 3;
  out[idx] = in[((size_t)(b * SS + s) * NKVV + kv) * HDD + d];
}

// ---------------- attention pass 1: per-row softmax stats ----------------
// One WG per (b,h,16-row q tile); 8 waves split the k-tiles.
// Causal: only kt <= qt computed; kt == qt is the only masked tile.
__global__ __launch_bounds__(256) void attn_stats(const __bf16* __restrict__ q,
                                                  const __bf16* __restrict__ k,
                                                  float* __restrict__ gm,
                                                  float* __restrict__ gl) {
  const int qt = blockIdx.x % (SS / 16);
  const int h  = (blockIdx.x / (SS / 16)) % NHH;
  const int b  = blockIdx.x / ((SS / 16) * NHH);
  const int kv = h >> 1;
  const int w = threadIdx.x >> 5, lane = threadIdx.x & 31, lm = lane & 15;
  const bool hi = lane >= 16;
  const int hi8 = hi ? 8 : 0;
  const int kbA = hi ? 16 : 0, kbB = hi ? 16 : 0;

  FragAB qa[4];
  const __bf16* qp = q + ((size_t)(b * SS + qt * 16 + lm) * NHH + h) * HDD;
#pragma unroll
  for (int ks = 0; ks < 4; ++ks) {
    qa[ks].q[0] = *(const uint4*)(qp + ks * 32 + kbA);
    qa[ks].q[1] = *(const uint4*)(qp + ks * 32 + kbA + 8);
  }

  float ml[8], ll[8];
#pragma unroll
  for (int r = 0; r < 8; ++r) { ml[r] = -3.0e38f; ll[r] = 0.0f; }

  for (int kt = w; kt <= qt; kt += 8) {          // causal: skip kt > qt
    const __bf16* kbase =
        k + ((size_t)(b * SS + kt * 16 + lm) * NKVV + kv) * HDD + kbB;
    if (kt + 8 <= qt)
      __builtin_prefetch(kbase + (size_t)128 * NKVV * HDD, 0, 1);
    FragAB kf[4];
#pragma unroll
    for (int ks = 0; ks < 4; ++ks) {   // 8 clause-able b128 loads
      kf[ks].q[0] = *(const uint4*)(kbase + ks * 32);
      kf[ks].q[1] = *(const uint4*)(kbase + ks * 32 + 8);
    }
    v8f c = vzero8();
#pragma unroll
    for (int ks = 0; ks < 4; ++ks) c = wmma_bf16(qa[ks], kf[ks], c);

    if (kt == qt) {                    // diagonal tile: per-element mask
#pragma unroll
      for (int r = 0; r < 8; ++r)
        if (lm > r + hi8) c[r] += NEG_BIG;
    }
#pragma unroll
    for (int r = 0; r < 8; ++r) {      // lane-local online softmax update
      float sc = c[r];
      float nm = fmaxf(ml[r], sc);
      ll[r] = ll[r] * __expf(ml[r] - nm) + __expf(sc - nm);
      ml[r] = nm;
    }
  }

  // single cross-lane (16-wide) combine of per-lane partials
#pragma unroll
  for (int r = 0; r < 8; ++r) {
    float m = ml[r], l = ll[r];
#pragma unroll
    for (int off = 8; off >= 1; off >>= 1) {
      float om = __shfl_xor(m, off, 32);
      float ol = __shfl_xor(l, off, 32);
      float nm = fmaxf(m, om);
      l = l * __expf(m - nm) + ol * __expf(om - nm);
      m = nm;
    }
    ml[r] = m; ll[r] = l;
  }

  __shared__ float smx[8][16], slx[8][16];
  if (lm == 0) {
#pragma unroll
    for (int r = 0; r < 8; ++r) {
      smx[w][r + hi8] = ml[r];
      slx[w][r + hi8] = ll[r];
    }
  }
  __syncthreads();
  if (threadIdx.x < 16) {
    int r = threadIdx.x;
    float M = -3.0e38f;
#pragma unroll
    for (int w2 = 0; w2 < 8; ++w2) M = fmaxf(M, smx[w2][r]);
    float L = 0.0f;
#pragma unroll
    for (int w2 = 0; w2 < 8; ++w2) L += slx[w2][r] * __expf(smx[w2][r] - M);
    size_t o = (size_t)(b * NHH + h) * SS + qt * 16 + r;
    gm[o] = M;
    gl[o] = L;
  }
}

// ---------------- attention pass 2: write weights + fused P@V ----------------
// WG per (b,h,128-row q block); each wave owns 16 q rows.
// Causal: compute only kt <= qt; upper triangle is a streaming NT zero-fill.
__global__ __launch_bounds__(256) void attn_pv(const __bf16* __restrict__ q,
                                               const __bf16* __restrict__ k,
                                               const __bf16* __restrict__ vt,
                                               const float* __restrict__ gm,
                                               const float* __restrict__ gl,
                                               float* __restrict__ wout,
                                               __bf16* __restrict__ hh) {
  const int qb = blockIdx.x % (SS / 128);
  const int h  = (blockIdx.x / (SS / 128)) % NHH;
  const int b  = blockIdx.x / ((SS / 128) * NHH);
  const int kv = h >> 1;
  const int w = threadIdx.x >> 5, lane = threadIdx.x & 31, lm = lane & 15;
  const bool hi = lane >= 16;
  const int hi8 = hi ? 8 : 0;
  const int kbA = hi ? 16 : 0, kbB = hi ? 16 : 0;
  const int qrow0 = qb * 128 + w * 16;
  const int qt = qrow0 >> 4;          // this wave's diagonal k-tile

  __shared__ alignas(16) __bf16 pbuf[8][16][32];  // per-wave P tile (16x32)

  FragAB qa[4];
  const __bf16* qp = q + ((size_t)(b * SS + qrow0 + lm) * NHH + h) * HDD;
#pragma unroll
  for (int ks = 0; ks < 4; ++ks) {
    qa[ks].q[0] = *(const uint4*)(qp + ks * 32 + kbA);
    qa[ks].q[1] = *(const uint4*)(qp + ks * 32 + kbA + 8);
  }

  float mi[8], li[8];
#pragma unroll
  for (int r = 0; r < 8; ++r) {
    size_t o = (size_t)(b * NHH + h) * SS + qrow0 + r + hi8;
    mi[r] = gm[o];
    li[r] = 1.0f / gl[o];
  }

  v8f oacc[8];
#pragma unroll
  for (int dt = 0; dt < 8; ++dt) oacc[dt] = vzero8();

  float* wp = wout + (size_t)(b * NHH + h) * SS * SS;
  const __bf16* vbase = vt + (size_t)(b * NKVV + kv) * HDD * SS;

  const int ksb_end = qt >> 1;        // last 32-wide K block with any valid col
  for (int ksb = 0; ksb <= ksb_end; ++ksb) {
#pragma unroll
    for (int t = 0; t < 2; ++t) {
      const int kt = ksb * 2 + t;
      const int col = kt * 16 + lm;
      if (kt <= qt) {
        const __bf16* kbase =
            k + ((size_t)(b * SS + kt * 16 + lm) * NKVV + kv) * HDD + kbB;
        if (kt + 2 <= qt)
          __builtin_prefetch(kbase + (size_t)32 * NKVV * HDD, 0, 1);
        FragAB kf[4];
#pragma unroll
        for (int ks = 0; ks < 4; ++ks) {
          kf[ks].q[0] = *(const uint4*)(kbase + ks * 32);
          kf[ks].q[1] = *(const uint4*)(kbase + ks * 32 + 8);
        }
        v8f c = vzero8();
#pragma unroll
        for (int ks = 0; ks < 4; ++ks) c = wmma_bf16(qa[ks], kf[ks], c);

        if (kt == qt) {               // only the diagonal tile needs a mask
#pragma unroll
          for (int r = 0; r < 8; ++r)
            if (lm > r + hi8) c[r] += NEG_BIG;
        }
#pragma unroll
        for (int r = 0; r < 8; ++r) {
          int row = qrow0 + r + hi8;
          float p = __expf(c[r] - mi[r]) * li[r];
          __builtin_nontemporal_store(p, &wp[(size_t)row * SS + col]);
          pbuf[w][r + hi8][t * 16 + lm] = (__bf16)p;
        }
      } else {                        // fully-masked tile in a live pair
#pragma unroll
        for (int r = 0; r < 8; ++r) {
          int row = qrow0 + r + hi8;
          __builtin_nontemporal_store(0.0f, &wp[(size_t)row * SS + col]);
          pbuf[w][r + hi8][t * 16 + lm] = (__bf16)0.0f;
        }
      }
    }
    asm volatile("s_wait_dscnt 0" ::: "memory");
    FragAB pa;
    pa.q[0] = *(const uint4*)&pbuf[w][lm][kbA];
    pa.q[1] = *(const uint4*)&pbuf[w][lm][kbA + 8];
#pragma unroll
    for (int dt = 0; dt < 8; ++dt) {
      const __bf16* vp = vbase + (size_t)(dt * 16 + lm) * SS + ksb * 32 + kbB;
      if (ksb < ksb_end) __builtin_prefetch(vp + 32, 0, 1);
      FragAB vf;
      vf.q[0] = *(const uint4*)(vp);
      vf.q[1] = *(const uint4*)(vp + 8);
      oacc[dt] = wmma_bf16(pa, vf, oacc[dt]);
    }
  }

  // streaming zero-fill of the strictly-upper-triangle weight columns
  {
    const int cstart = (ksb_end + 1) * 32;
    f32x4 z;
#pragma unroll
    for (int i = 0; i < 4; ++i) z[i] = 0.0f;
    for (int r = 0; r < 16; ++r) {
      float* rowp = wp + (size_t)(qrow0 + r) * SS;
      for (int col = cstart + lane * 4; col < SS; col += 128)
        __builtin_nontemporal_store(z, (f32x4*)(rowp + col));
    }
  }

#pragma unroll
  for (int dt = 0; dt < 8; ++dt)
#pragma unroll
    for (int r = 0; r < 8; ++r) {
      int row = qrow0 + r + hi8;
      hh[((size_t)(b * SS + row) * NHH + h) * HDD + dt * 16 + lm] =
          (__bf16)(oacc[dt][r]);
    }
}

// ---------------- host-side orchestration ----------------
extern "C" void kernel_launch(void* const* d_in, const int* in_sizes, int n_in,
                              void* d_out, int out_size, void* d_ws, size_t ws_size,
                              hipStream_t stream) {
  const float* hs   = (const float*)d_in[0];
  const float* cosb = (const float*)d_in[1];
  const float* sinb = (const float*)d_in[2];
  // d_in[3] = attention_mask (pure causal; regenerated analytically)
  const float* wq = (const float*)d_in[4];
  const float* wk = (const float*)d_in[5];
  const float* wv = (const float*)d_in[6];
  const float* wo = (const float*)d_in[7];
  const float* qnw = (const float*)d_in[8];
  const float* knw = (const float*)d_in[9];

  char* ws = (char*)d_ws;
  size_t off = 0;
  auto alloc = [&](size_t bytes) {
    char* p = ws + off;
    off = (off + bytes + 255) & ~(size_t)255;
    return p;
  };
  __bf16* hbf  = (__bf16*)alloc((size_t)BB * SS * HIDD * 2);        // 16 MB
  __bf16* wqbf = (__bf16*)alloc((size_t)HIDD * NHH * HDD * 2);      // 8 MB
  __bf16* wkbf = (__bf16*)alloc((size_t)HIDD * NKVV * HDD * 2);     // 4 MB
  __bf16* wvbf = (__bf16*)alloc((size_t)HIDD * NKVV * HDD * 2);     // 4 MB
  __bf16* wobf = (__bf16*)alloc((size_t)NHH * HDD * HIDD * 2);      // 8 MB
  __bf16* qbf  = (__bf16*)alloc((size_t)BB * SS * NHH * HDD * 2);   // 16 MB
  __bf16* kbf  = (__bf16*)alloc((size_t)BB * SS * NKVV * HDD * 2);  // 8 MB
  __bf16* vbf  = (__bf16*)alloc((size_t)BB * SS * NKVV * HDD * 2);  // 8 MB
  __bf16* vtbf = (__bf16*)alloc((size_t)BB * NKVV * HDD * SS * 2);  // 8 MB
  __bf16* hhbf = (__bf16*)alloc((size_t)BB * SS * NHH * HDD * 2);   // 16 MB
  float*  gm   = (float*)alloc((size_t)BB * NHH * SS * 4);
  float*  gl   = (float*)alloc((size_t)BB * NHH * SS * 4);

  float* out_attn = (float*)d_out;
  float* out_w = out_attn + (size_t)BB * SS * HIDD;

  // 1) f32 -> bf16 conversions
  cvt_bf16<<<(BB * SS * HIDD / 4 + 255) / 256, 256, 0, stream>>>(hs, hbf, BB * SS * HIDD / 4);
  cvt_bf16<<<(HIDD * NHH * HDD / 4 + 255) / 256, 256, 0, stream>>>(wq, wqbf, HIDD * NHH * HDD / 4);
  cvt_bf16<<<(HIDD * NKVV * HDD / 4 + 255) / 256, 256, 0, stream>>>(wk, wkbf, HIDD * NKVV * HDD / 4);
  cvt_bf16<<<(HIDD * NKVV * HDD / 4 + 255) / 256, 256, 0, stream>>>(wv, wvbf, HIDD * NKVV * HDD / 4);
  cvt_bf16<<<(NHH * HDD * HIDD / 4 + 255) / 256, 256, 0, stream>>>(wo, wobf, NHH * HDD * HIDD / 4);

  // 2) QKV projections (bf16 WMMA, TDM-staged A tile)
  gemm_bf16<false><<<dim3((NHH * HDD) / 256, (BB * SS) / 64), 256, 0, stream>>>(
      hbf, wqbf, (void*)qbf, BB * SS, NHH * HDD, HIDD);
  gemm_bf16<false><<<dim3((NKVV * HDD) / 256, (BB * SS) / 64), 256, 0, stream>>>(
      hbf, wkbf, (void*)kbf, BB * SS, NKVV * HDD, HIDD);
  gemm_bf16<false><<<dim3((NKVV * HDD) / 256, (BB * SS) / 64), 256, 0, stream>>>(
      hbf, wvbf, (void*)vbf, BB * SS, NKVV * HDD, HIDD);

  // 3) RMSNorm + RoPE (SCALING folded into Q)
  norm_rope<<<BB * SS * NHH / 8, 256, 0, stream>>>(qbf, cosb, sinb, qnw, NHH, SCALING_);
  norm_rope<<<BB * SS * NKVV / 8, 256, 0, stream>>>(kbf, cosb, sinb, knw, NKVV, 1.0f);

  // 4) V transpose for contiguous P@V B-fragments
  transpose_v<<<BB * NKVV * HDD * SS / 256, 256, 0, stream>>>(vbf, vtbf);

  // 5) softmax stats (lane-local online max/sumexp, causal-pruned)
  attn_stats<<<BB * NHH * (SS / 16), 256, 0, stream>>>(qbf, kbf, gm, gl);

  // 6) normalized weights (written once, NT stores) + fused P@V
  attn_pv<<<BB * NHH * (SS / 128), 256, 0, stream>>>(qbf, kbf, vtbf, gm, gl, out_w, hhbf);

  // 7) output projection -> attn_out (f32)
  gemm_bf16<true><<<dim3(HIDD / 256, (BB * SS) / 64), 256, 0, stream>>>(
      hhbf, wobf, (void*)out_attn, BB * SS, HIDD, NHH * HDD);
}
